// GCN_layer_1460288881131
// MI455X (gfx1250) — matrix operand order repaired and checked by reading, not compiled, not run on hardware
//
#include <hip/hip_runtime.h>

// ---------------------------------------------------------------------------
// GCN layer: h = x@W + b ; out = relu(adj@h + x)
// B=32, N=1024, F=128.  bf16 WMMA (v_wmma_f32_16x16x32_bf16), f32 accum.
// ---------------------------------------------------------------------------

#define BATCH 32
#define NNODE 1024
#define FDIM  128
#define KSTEP 32
#define LDS_STRIDE 40   // halves per row (32 data + 8 pad) => 80B, 16B aligned

typedef __attribute__((ext_vector_type(16))) __bf16 v16bf;
typedef __attribute__((ext_vector_type(8)))  float  v8f;

struct __attribute__((aligned(16))) U4 { unsigned int x, y, z, w; };

union AFrag { v16bf v; U4 q[2]; };

// f32 pair -> packed bf16 dword.  Prefer the HW converter; fallback is
// round-to-nearest + v_perm_b32 byte pack (3 VALU per pair).
#if __has_builtin(__builtin_amdgcn_cvt_pk_bf16_f32)
__device__ __forceinline__ unsigned int pk2(float a, float b) {
    auto r = __builtin_amdgcn_cvt_pk_bf16_f32(a, b);   // D[15:0]=bf16(a), D[31:16]=bf16(b)
    return __builtin_bit_cast(unsigned int, r);
}
#else
__device__ __forceinline__ unsigned int pk2(float a, float b) {
    unsigned int ua = __float_as_uint(a) + 0x8000u;    // round to nearest
    unsigned int ub = __float_as_uint(b) + 0x8000u;
    // result bytes [0,1] = ua[2,3]; bytes [2,3] = ub[2,3]  (v_perm_b32)
    return __builtin_amdgcn_perm(ub, ua, 0x07060302u);
}
#endif

__device__ __forceinline__ U4 pack16(const float4& f0, const float4& f1) {
    U4 q;
    q.x = pk2(f0.x, f0.y); q.y = pk2(f0.z, f0.w);
    q.z = pk2(f1.x, f1.y); q.w = pk2(f1.z, f1.w);
    return q;
}

// Async global->LDS 16B copy (CDNA5 data mover path, ASYNCcnt tracked).
__device__ __forceinline__ void async_copy_b128(unsigned lds_off, const void* gptr) {
    asm volatile("global_load_async_to_lds_b128 %0, %1, off"
                 :
                 : "v"(lds_off), "v"((unsigned long long)(size_t)gptr)
                 : "memory");
}

__device__ __forceinline__ void wait_async0() {
#if __has_builtin(__builtin_amdgcn_s_wait_asynccnt)
    __builtin_amdgcn_s_wait_asynccnt(0);
#else
    asm volatile("s_wait_asynccnt 0x0" ::: "memory");
#endif
}

// ---------------------------------------------------------------------------
// Kernel 1: h_sw[b][m>>5][f][m&31] = bf16( x[b][m][:] @ W[:][f] + bias[f] )
// grid = BATCH * (NNODE/128) blocks, 256 threads (8 waves)
// ---------------------------------------------------------------------------
__global__ void __launch_bounds__(256)
gcn_gemm1_xw(const float* __restrict__ x, const float* __restrict__ W,
             const float* __restrict__ bias, unsigned short* __restrict__ h_sw) {
    __shared__ __attribute__((aligned(16))) unsigned short sA[128 * LDS_STRIDE];
    __shared__ __attribute__((aligned(16))) unsigned short sB[128 * LDS_STRIDE];

    const int tid   = threadIdx.x;
    const int wave  = tid >> 5;
    const int lane  = tid & 31;
    const int lh    = lane >> 4;   // half-wave id
    const int l15   = lane & 15;

    const int b     = blockIdx.x >> 3;
    const int rbase = (blockIdx.x & 7) * 128;

    v8f acc[8] = {};

    const int lrow  = tid >> 1;    // 0..127
    const int lhalf = tid & 1;     // 0..1 (16 k-values each)

#pragma unroll
    for (int kb = 0; kb < FDIM / KSTEP; ++kb) {
        // ---- stage A = x tile [128 m][32 k] as bf16 -------------------------
        {
            const float4* src = reinterpret_cast<const float4*>(
                x + (size_t)((b * NNODE + rbase + lrow) * FDIM + kb * KSTEP + lhalf * 16));
            float4 f0 = src[0], f1 = src[1], f2 = src[2], f3 = src[3];
            U4* dst = reinterpret_cast<U4*>(&sA[lrow * LDS_STRIDE + lhalf * 16]);
            dst[0] = pack16(f0, f1);
            dst[1] = pack16(f2, f3);
        }
        // ---- stage B = W^T tile [128 f][32 k] as bf16 -----------------------
        {
            const int f  = lrow;
            const int k0 = kb * KSTEP + lhalf * 16;
            float v[16];
#pragma unroll
            for (int j = 0; j < 16; ++j) v[j] = W[(k0 + j) * FDIM + f];
            U4 q0, q1;
            q0.x = pk2(v[0], v[1]);   q0.y = pk2(v[2], v[3]);
            q0.z = pk2(v[4], v[5]);   q0.w = pk2(v[6], v[7]);
            q1.x = pk2(v[8], v[9]);   q1.y = pk2(v[10], v[11]);
            q1.z = pk2(v[12], v[13]); q1.w = pk2(v[14], v[15]);
            U4* dst = reinterpret_cast<U4*>(&sB[f * LDS_STRIDE + lhalf * 16]);
            dst[0] = q0;
            dst[1] = q1;
        }
        __syncthreads();

        // ---- WMMA: wave w does rows [w*16, w*16+16), all 8 col tiles --------
        AFrag fa;
        const int aoff = (wave * 16 + l15) * LDS_STRIDE + lh * 8;
        fa.q[0] = *reinterpret_cast<const U4*>(&sA[aoff]);
        fa.q[1] = *reinterpret_cast<const U4*>(&sA[aoff + 16]);
#pragma unroll
        for (int t = 0; t < 8; ++t) {
            AFrag fb;
            const int boff = (t * 16 + l15) * LDS_STRIDE + lh * 8;
            fb.q[0] = *reinterpret_cast<const U4*>(&sB[boff]);
            fb.q[1] = *reinterpret_cast<const U4*>(&sB[boff + 16]);
            acc[t] = __builtin_amdgcn_wmma_f32_16x16x32_bf16(
                false, fa.v, false, fb.v, (short)0, acc[t], false, false);
        }
        __syncthreads();
    }

    // ---- epilogue: +bias, write pre-swizzled bf16 h ------------------------
    // lane element r: m = rbase + wave*16 + r + 8*lh, f = t*16 + l15
    // h_sw index = (b*32 + m>>5)*4096 + f*32 + (m&31); r=0..7 are contiguous.
    const int slab = b * (NNODE / 32) + (rbase >> 5) + (wave >> 1);
    const int m0i  = (wave & 1) * 16 + lh * 8;   // m&31 for r=0
#pragma unroll
    for (int t = 0; t < 8; ++t) {
        const int fcol = t * 16 + l15;
        const float bv = bias[fcol];
        U4 q;
        q.x = pk2(acc[t][0] + bv, acc[t][1] + bv);
        q.y = pk2(acc[t][2] + bv, acc[t][3] + bv);
        q.z = pk2(acc[t][4] + bv, acc[t][5] + bv);
        q.w = pk2(acc[t][6] + bv, acc[t][7] + bv);
        *reinterpret_cast<U4*>(h_sw + (size_t)slab * 4096 + fcol * 32 + m0i) = q;
    }
}

// ---------------------------------------------------------------------------
// Kernel 2: out = relu( adj @ h + x )
// grid = BATCH * (NNODE/128) blocks, 256 threads (8 waves)
// ---------------------------------------------------------------------------
__global__ void __launch_bounds__(256)
gcn_gemm2_agg(const float* __restrict__ adj, const unsigned short* __restrict__ h_sw,
              const float* __restrict__ x, float* __restrict__ out) {
    __shared__ __attribute__((aligned(16))) unsigned short sA[128 * LDS_STRIDE];
    __shared__ __attribute__((aligned(16))) unsigned short sB[128 * LDS_STRIDE];

    const int tid   = threadIdx.x;
    const int wave  = tid >> 5;
    const int lane  = tid & 31;
    const int lh    = lane >> 4;
    const int l15   = lane & 15;

    const int b     = blockIdx.x >> 3;
    const int rbase = (blockIdx.x & 7) * 128;

    v8f acc[8] = {};

    const int lrow  = tid >> 1;
    const int lhalf = tid & 1;
    const float* arow = adj + (size_t)(b * NNODE + rbase + lrow) * NNODE + lhalf * 16;

    // Per-thread LDS destinations for the async B-slab copy (2 x 16B each).
    const int c0 = tid, c1 = tid + 256;
    const unsigned ldsB0 =
        (unsigned)(size_t)(void*)&sB[(c0 >> 2) * LDS_STRIDE + (c0 & 3) * 8];
    const unsigned ldsB1 =
        (unsigned)(size_t)(void*)&sB[(c1 >> 2) * LDS_STRIDE + (c1 & 3) * 8];

    for (int kb = 0; kb < NNODE / KSTEP; ++kb) {
        // ---- stage B = h_sw slab [128 f][32 k]: async global->LDS copy ------
        {
            const U4* src = reinterpret_cast<const U4*>(
                h_sw + (size_t)(b * (NNODE / 32) + kb) * 4096);
            async_copy_b128(ldsB0, src + c0);
            async_copy_b128(ldsB1, src + c1);
        }
        // ---- stage A = adj tile [128 n][32 m], fp32 -> bf16 -----------------
        {
            const float4* src = reinterpret_cast<const float4*>(arow + kb * KSTEP);
            float4 f0 = src[0], f1 = src[1], f2 = src[2], f3 = src[3];
            if (kb + 1 < NNODE / KSTEP)
                __builtin_prefetch(arow + (kb + 1) * KSTEP, 0, 0);  // global_prefetch_b8
            U4* dst = reinterpret_cast<U4*>(&sA[lrow * LDS_STRIDE + lhalf * 16]);
            dst[0] = pack16(f0, f1);
            dst[1] = pack16(f2, f3);
        }
        wait_async0();          // our async copies landed in LDS
        __syncthreads();        // (implied dscnt wait covers the A-tile stores)

        // ---- WMMA ----------------------------------------------------------
        AFrag fa;
        const int aoff = (wave * 16 + l15) * LDS_STRIDE + lh * 8;
        fa.q[0] = *reinterpret_cast<const U4*>(&sA[aoff]);
        fa.q[1] = *reinterpret_cast<const U4*>(&sA[aoff + 16]);
#pragma unroll
        for (int t = 0; t < 8; ++t) {
            AFrag fb;
            const int boff = (t * 16 + l15) * LDS_STRIDE + lh * 8;
            fb.q[0] = *reinterpret_cast<const U4*>(&sB[boff]);
            fb.q[1] = *reinterpret_cast<const U4*>(&sB[boff + 16]);
            acc[t] = __builtin_amdgcn_wmma_f32_16x16x32_bf16(
                false, fa.v, false, fb.v, (short)0, acc[t], false, false);
        }
        __syncthreads();
    }

    // ---- epilogue: residual + ReLU, fp32 out -------------------------------
    const int nrow0 = rbase + wave * 16 + lh * 8;   // row for r=0
#pragma unroll
    for (int t = 0; t < 8; ++t) {
        const int fcol = t * 16 + l15;
#pragma unroll
        for (int r = 0; r < 8; ++r) {
            const size_t idx = ((size_t)b * NNODE + nrow0 + r) * FDIM + fcol;
            float v = acc[t][r] + x[idx];
            out[idx] = v > 0.0f ? v : 0.0f;
        }
    }
}

// ---------------------------------------------------------------------------
extern "C" void kernel_launch(void* const* d_in, const int* in_sizes, int n_in,
                              void* d_out, int out_size, void* d_ws, size_t ws_size,
                              hipStream_t stream) {
    (void)in_sizes; (void)n_in; (void)out_size; (void)ws_size;
    const float* x    = (const float*)d_in[0];   // [32,1024,128]
    const float* adj  = (const float*)d_in[1];   // [32,1024,1024]
    const float* W    = (const float*)d_in[2];   // [128,128]
    const float* bias = (const float*)d_in[3];   // [128]
    float* out = (float*)d_out;                  // [32,1024,128]
    unsigned short* h_sw = (unsigned short*)d_ws; // 8 MB bf16 swizzled h

    dim3 grid(BATCH * (NNODE / 128));
    dim3 block(256);
    gcn_gemm1_xw<<<grid, block, 0, stream>>>(x, W, bias, h_sw);
    gcn_gemm2_agg<<<grid, block, 0, stream>>>(adj, h_sw, x, out);
}
